// Model_2740189135128
// MI455X (gfx1250) — compile-verified, matrix operand order checked by reading, not run on hardware
//
#include <hip/hip_runtime.h>
#include <hip/hip_bf16.h>
#include <math.h>

// ---------------------------------------------------------------------------
// CDNA5 (gfx1250) GRU seq2seq w/ attention.
// All matmuls use V_WMMA_F32_16X16X32_BF16 (bf16 in, f32 accumulate).
// Recurrent steps fuse h@Whh^T + GRU gates into a single WMMA kernel.
// ---------------------------------------------------------------------------

typedef __attribute__((ext_vector_type(16))) __bf16 v16bf;
typedef __attribute__((ext_vector_type(8)))  float  v8f;

// Fixed problem dims (from the reference).
constexpr int B  = 32;
constexpr int SE = 256;
constexpr int SD = 64;
constexpr int E  = 512;
constexpr int H  = 512;
constexpr int H3 = 3 * H;    // 1536 encoder gates
constexpr int H2 = 2 * H;    // 1024 decoder hidden / enc feature dim
constexpr int G2 = 3 * H2;   // 3072 decoder gates
constexpr int AQ = 4 * H;    // 2048 attn+query concat
constexpr int C  = 32000;    // vocab

// ---------------------------- device helpers -------------------------------
__device__ __forceinline__ unsigned short f2bf(float f) {
  unsigned u = __float_as_uint(f);
  u += 0x7fffu + ((u >> 16) & 1u);          // round-to-nearest-even
  return (unsigned short)(u >> 16);
}
__device__ __forceinline__ float bf2f(unsigned short h) {
  return __uint_as_float(((unsigned)h) << 16);
}
__device__ __forceinline__ float sigm(float x) { return 1.0f / (1.0f + __expf(-x)); }

__device__ __forceinline__ v8f wmma_bf16(v16bf a, v16bf b, v8f c) {
  return __builtin_amdgcn_wmma_f32_16x16x32_bf16(false, a, false, b,
                                                 (short)0, c, false, false);
}

// ---------------------------- WMMA GEMM ------------------------------------
// D[M,N] = A[M,K] @ W[N,K]^T (+bias[N]), optional tanh epilogue.
// One wave computes a 32(M) x 16*NT(N) strip:
//   grid.x = N/(16*NT), grid.y = M/32, block = 32 threads (1 wave).
// bf16 fragment, this code's k order (same for A and B): row = lane&15, K-half = (lane>>4)*16,
// 16 contiguous bf16 per lane. W stored (N,K) row-major => B-operand loads
// use identical addressing on W rows.
template <int NT, int TANH_EPI>
__global__ __launch_bounds__(32) void wmma_gemm(
    const unsigned short* __restrict__ A, int lda,
    const unsigned short* __restrict__ W, int ldw,
    const float* __restrict__ bias,
    float* __restrict__ Df, int lddf,
    unsigned short* __restrict__ Dh, int lddh,
    int K) {
  const int lane = threadIdx.x;
  const int n0 = blockIdx.x * (16 * NT);
  const int m0 = blockIdx.y * 32;
  const int r  = lane & 15;
  const int kh = (lane >> 4) << 4;

  const unsigned short* ap0 = A + (size_t)(m0 + r) * lda + kh;
  const unsigned short* ap1 = ap0 + (size_t)16 * lda;
  const unsigned short* wp[NT];
#pragma unroll
  for (int t = 0; t < NT; ++t)
    wp[t] = W + (size_t)(n0 + 16 * t + r) * ldw + kh;

  v8f acc0[NT];
  v8f acc1[NT];
#pragma unroll
  for (int t = 0; t < NT; ++t) {
    v8f z = {};
    acc0[t] = z;
    acc1[t] = z;
  }

#pragma unroll 2
  for (int k = 0; k < K; k += 32) {
    v16bf a0 = *reinterpret_cast<const v16bf*>(ap0 + k);
    v16bf a1 = *reinterpret_cast<const v16bf*>(ap1 + k);
#pragma unroll
    for (int t = 0; t < NT; ++t) {
      v16bf w = *reinterpret_cast<const v16bf*>(wp[t] + k);
      acc0[t] = wmma_bf16(a0, w, acc0[t]);
      acc1[t] = wmma_bf16(a1, w, acc1[t]);
    }
  }

  const int rowBase = m0 + ((lane >> 4) << 3);  // C/D: vgpr i -> M = i + 8*(lane>>4)
#pragma unroll
  for (int t = 0; t < NT; ++t) {
    const int col = n0 + 16 * t + (lane & 15);
    const float bv = bias ? bias[col] : 0.0f;
#pragma unroll
    for (int i = 0; i < 8; ++i) {
      float v0 = acc0[t][i] + bv;
      float v1 = acc1[t][i] + bv;
      if (TANH_EPI) { v0 = tanhf(v0); v1 = tanhf(v1); }
      const int row0 = rowBase + i;
      const int row1 = rowBase + 16 + i;
      if (Df) {
        Df[(size_t)row0 * lddf + col] = v0;
        Df[(size_t)row1 * lddf + col] = v1;
      }
      if (Dh) {
        Dh[(size_t)row0 * lddh + col] = f2bf(v0);
        Dh[(size_t)row1 * lddh + col] = f2bf(v1);
      }
    }
  }
}

// ---------------------- fused GRU recurrent step ---------------------------
// One wave handles a 16-column slice j0..j0+15 of the hidden state for all
// B=32 batch rows: computes gh_{r,z,n} = h @ Whh[{0,1,2}*HH + j]^T via 6 WMMA
// accumulator chains, then applies the full GRU gate math in the epilogue.
//   grid.x = HH/16, block = 32 (1 wave).
// hprev_bf (read, whole rows) and hnext_bf (written) MUST be distinct buffers
// (cross-block WAR hazard) -> caller ping-pongs them.
template <int HH>
__global__ __launch_bounds__(32) void gru_hstep(
    const unsigned short* __restrict__ hprev_bf,  // (B, HH) bf16
    const unsigned short* __restrict__ Whh,       // (3*HH, HH) bf16
    const float* __restrict__ bhh,                // (3*HH)
    const float* __restrict__ gi,                 // gi base (+t offset applied)
    long gi_bstride,                              // elements between batch rows
    float* __restrict__ hstate,                   // (B, HH) f32 in/out
    unsigned short* __restrict__ hnext_bf,        // (B, HH) bf16 out
    unsigned short* __restrict__ enc_t,           // nullable, +t*2H +colOff applied
    long encB) {                                  // enc batch stride (elements)
  const int lane = threadIdx.x;
  const int j0 = blockIdx.x * 16;
  const int r  = lane & 15;
  const int kh = (lane >> 4) << 4;

  const unsigned short* ap0 = hprev_bf + (size_t)r * HH + kh;         // b = 0..15
  const unsigned short* ap1 = ap0 + (size_t)16 * HH;                  // b = 16..31
  const unsigned short* wr = Whh + (size_t)(j0 + r) * HH + kh;
  const unsigned short* wz = wr + (size_t)HH * HH;
  const unsigned short* wn = wz + (size_t)HH * HH;

  v8f a_r = {}, a_z = {}, a_n = {};   // batch rows 0..15
  v8f b_r = {}, b_z = {}, b_n = {};   // batch rows 16..31
#pragma unroll 2
  for (int k = 0; k < HH; k += 32) {
    v16bf h0 = *reinterpret_cast<const v16bf*>(ap0 + k);
    v16bf h1 = *reinterpret_cast<const v16bf*>(ap1 + k);
    v16bf fr = *reinterpret_cast<const v16bf*>(wr + k);
    v16bf fz = *reinterpret_cast<const v16bf*>(wz + k);
    v16bf fn = *reinterpret_cast<const v16bf*>(wn + k);
    a_r = wmma_bf16(h0, fr, a_r);
    b_r = wmma_bf16(h1, fr, b_r);
    a_z = wmma_bf16(h0, fz, a_z);
    b_z = wmma_bf16(h1, fz, b_z);
    a_n = wmma_bf16(h0, fn, a_n);
    b_n = wmma_bf16(h1, fn, b_n);
  }

  const int col = j0 + (lane & 15);
  const float bhr = bhh[col];
  const float bhz = bhh[HH + col];
  const float bhn = bhh[2 * HH + col];
  const int rowBase = (lane >> 4) << 3;

  auto emit = [&](int b, float ghr, float ghz, float ghn) {
    const float* gib = gi + (size_t)b * gi_bstride;
    float rr = sigm(gib[col] + ghr + bhr);
    float zz = sigm(gib[HH + col] + ghz + bhz);
    float nn = tanhf(gib[2 * HH + col] + rr * (ghn + bhn));
    float hv = hstate[(size_t)b * HH + col];
    float hnew = (1.0f - zz) * nn + zz * hv;
    hstate[(size_t)b * HH + col] = hnew;
    unsigned short h16 = f2bf(hnew);
    hnext_bf[(size_t)b * HH + col] = h16;
    if (enc_t) enc_t[(size_t)b * encB + col] = h16;
  };
#pragma unroll
  for (int i = 0; i < 8; ++i) {
    emit(rowBase + i,      a_r[i], a_z[i], a_n[i]);
    emit(rowBase + 16 + i, b_r[i], b_z[i], b_n[i]);
  }
}

// ---------------------------- small kernels --------------------------------
__global__ __launch_bounds__(256) void conv_f32_bf16(const float* __restrict__ s,
                                                     unsigned short* __restrict__ d,
                                                     int n) {
  int i = blockIdx.x * 256 + threadIdx.x;
  if (i < n) d[i] = f2bf(s[i]);
}
__global__ __launch_bounds__(256) void zero_f32(float* p, int n) {
  int i = blockIdx.x * 256 + threadIdx.x;
  if (i < n) p[i] = 0.0f;
}
__global__ __launch_bounds__(256) void zero_u16(unsigned short* p, int n) {
  int i = blockIdx.x * 256 + threadIdx.x;
  if (i < n) p[i] = 0;
}

// decin[:, 0:E] = outp[:, t, :]   (op half written by tanh-GEMM of prev step)
__global__ __launch_bounds__(256) void build_decin(
    const unsigned short* __restrict__ outp_h,  // (B, SD, E)
    int t,
    unsigned short* __restrict__ decin) {       // (B, 2E)
  int idx = blockIdx.x * 256 + threadIdx.x;     // B*E
  int b = idx >> 9;
  int j = idx & (E - 1);
  decin[(size_t)b * (2 * E) + j] = outp_h[((size_t)b * SD + t) * E + j];
}

// h_d = concat(h_fwd_final, h_bwd_final)
__global__ __launch_bounds__(256) void build_hd(const float* __restrict__ hf,
                                                const float* __restrict__ hb,
                                                float* __restrict__ hd,
                                                unsigned short* __restrict__ hd_bf) {
  int idx = blockIdx.x * 256 + threadIdx.x;   // B*H2
  int b = idx >> 10;
  int j = idx & (H2 - 1);
  float v = (j < H) ? hf[b * H + j] : hb[b * H + (j - H)];
  hd[idx] = v;
  hd_bf[idx] = f2bf(v);
}

// scores[b,s] = sum_h enc[b,s,h] * q[b,h]
__global__ __launch_bounds__(256) void attn_scores(
    const unsigned short* __restrict__ enc,  // (B, SE, H2)
    const float* __restrict__ q,             // (B, H2)
    float* __restrict__ scores) {            // (B, SE)
  int idx = blockIdx.x * 256 + threadIdx.x;  // B*SE = 8192
  int b = idx >> 8;
  int s = idx & (SE - 1);
  const unsigned short* e = enc + ((size_t)b * SE + s) * H2;
  const float* qb = q + (size_t)b * H2;
  float acc = 0.0f;
#pragma unroll 4
  for (int h = 0; h < H2; h += 2) {
    unsigned pk = *reinterpret_cast<const unsigned*>(e + h);
    acc += bf2f((unsigned short)(pk & 0xffffu)) * qb[h];
    acc += bf2f((unsigned short)(pk >> 16)) * qb[h + 1];
  }
  scores[idx] = acc;
}

// softmax over s (one block per batch row); writes aw and the attns output.
__global__ __launch_bounds__(SE) void softmax_attn(
    const float* __restrict__ scores, float* __restrict__ aw,
    float* __restrict__ attns_out) {
  __shared__ float red[SE];
  int b = blockIdx.x;
  int s = threadIdx.x;
  float v = scores[b * SE + s];
  red[s] = v;
  __syncthreads();
  for (int off = SE / 2; off > 0; off >>= 1) {
    if (s < off) red[s] = fmaxf(red[s], red[s + off]);
    __syncthreads();
  }
  float m = red[0];
  __syncthreads();
  float e = __expf(v - m);
  red[s] = e;
  __syncthreads();
  for (int off = SE / 2; off > 0; off >>= 1) {
    if (s < off) red[s] += red[s + off];
    __syncthreads();
  }
  float w = e / red[0];
  aw[b * SE + s] = w;
  attns_out[b * SE + s] = w;
}

// attnq[:, 0:H2] = sum_s aw[b,s] * enc[b,s,:]
__global__ __launch_bounds__(256) void attn_combine(
    const float* __restrict__ aw,            // (B, SE)
    const unsigned short* __restrict__ enc,  // (B, SE, H2)
    unsigned short* __restrict__ attnq) {    // (B, AQ)
  int idx = blockIdx.x * 256 + threadIdx.x;  // B*H2
  int b = idx >> 10;
  int j = idx & (H2 - 1);
  const float* w = aw + (size_t)b * SE;
  const unsigned short* e = enc + (size_t)b * SE * H2 + j;
  float acc = 0.0f;
#pragma unroll 4
  for (int s = 0; s < SE; ++s) acc += w[s] * bf2f(e[(size_t)s * H2]);
  attnq[(size_t)b * AQ + j] = f2bf(acc);
}

// ---------------------------- host side ------------------------------------
static inline void launch_gemm(hipStream_t st, int NT, const unsigned short* A,
                               int lda, const unsigned short* W, int ldw,
                               const float* bias, float* Df, int lddf,
                               unsigned short* Dh, int lddh, int M, int N, int K,
                               bool tanh_epi) {
  dim3 g((unsigned)(N / (16 * NT)), (unsigned)(M / 32));
  if (NT == 4) {
    wmma_gemm<4, 0><<<g, 32, 0, st>>>(A, lda, W, ldw, bias, Df, lddf, Dh, lddh, K);
  } else if (NT == 2) {
    wmma_gemm<2, 0><<<g, 32, 0, st>>>(A, lda, W, ldw, bias, Df, lddf, Dh, lddh, K);
  } else if (tanh_epi) {
    wmma_gemm<1, 1><<<g, 32, 0, st>>>(A, lda, W, ldw, bias, Df, lddf, Dh, lddh, K);
  } else {
    wmma_gemm<1, 0><<<g, 32, 0, st>>>(A, lda, W, ldw, bias, Df, lddf, Dh, lddh, K);
  }
}
static inline void launch_conv(hipStream_t st, const float* s, unsigned short* d,
                               size_t n) {
  conv_f32_bf16<<<(unsigned)((n + 255) / 256), 256, 0, st>>>(s, d, (int)n);
}

extern "C" void kernel_launch(void* const* d_in, const int* in_sizes, int n_in,
                              void* d_out, int out_size, void* d_ws, size_t ws_size,
                              hipStream_t stream) {
  (void)in_sizes; (void)n_in; (void)out_size; (void)ws_size;

  const float* inp       = (const float*)d_in[0];
  const float* outp      = (const float*)d_in[1];
  const float* eWih_f    = (const float*)d_in[2];
  const float* eWhh_f    = (const float*)d_in[3];
  const float* ebih_f    = (const float*)d_in[4];
  const float* ebhh_f    = (const float*)d_in[5];
  const float* eWih_b    = (const float*)d_in[6];
  const float* eWhh_b    = (const float*)d_in[7];
  const float* ebih_b    = (const float*)d_in[8];
  const float* ebhh_b    = (const float*)d_in[9];
  const float* dWih      = (const float*)d_in[10];
  const float* dWhh      = (const float*)d_in[11];
  const float* dbih      = (const float*)d_in[12];
  const float* dbhh      = (const float*)d_in[13];
  const float* attn_in_W = (const float*)d_in[14];
  const float* attn_out_W= (const float*)d_in[15];
  const float* gen_W     = (const float*)d_in[16];
  const float* gen_b     = (const float*)d_in[17];

  float* outF = (float*)d_out;
  float* attnsBase = outF + (size_t)SD * B * C;   // outputs first, attns second

  // ---- workspace carve-up (256B-aligned bump allocator) ----
  char* p = (char*)d_ws;
  auto alloc = [&](size_t bytes) -> void* {
    void* r = (void*)p;
    p += (bytes + 255) & ~(size_t)255;
    return r;
  };
  auto aU16 = [&](size_t n) { return (unsigned short*)alloc(n * 2); };
  auto aF32 = [&](size_t n) { return (float*)alloc(n * 4); };

  unsigned short* inp_h   = aU16((size_t)B * SE * E);
  unsigned short* outp_h  = aU16((size_t)B * SD * E);
  unsigned short* eWih_fh = aU16((size_t)H3 * E);
  unsigned short* eWhh_fh = aU16((size_t)H3 * H);
  unsigned short* eWih_bh = aU16((size_t)H3 * E);
  unsigned short* eWhh_bh = aU16((size_t)H3 * H);
  unsigned short* dWih_h  = aU16((size_t)G2 * H2);
  unsigned short* dWhh_h  = aU16((size_t)G2 * H2);
  unsigned short* ain_h   = aU16((size_t)H2 * H2);
  unsigned short* aout_h  = aU16((size_t)E * AQ);
  unsigned short* genW_h  = aU16((size_t)C * AQ);
  unsigned short* enc_h   = aU16((size_t)B * SE * H2);
  unsigned short* hf_h0   = aU16((size_t)B * H);
  unsigned short* hf_h1   = aU16((size_t)B * H);
  unsigned short* hb_h0   = aU16((size_t)B * H);
  unsigned short* hb_h1   = aU16((size_t)B * H);
  unsigned short* hd_h0   = aU16((size_t)B * H2);
  unsigned short* hd_h1   = aU16((size_t)B * H2);
  unsigned short* decin_h = aU16((size_t)B * 2 * E);
  unsigned short* attnq_h = aU16((size_t)B * AQ);

  float* gi_f    = aF32((size_t)B * SE * H3);
  float* gi_b    = aF32((size_t)B * SE * H3);
  float* hf_f32  = aF32((size_t)B * H);
  float* hb_f32  = aF32((size_t)B * H);
  float* hd_f32  = aF32((size_t)B * H2);
  float* gi_d    = aF32((size_t)B * G2);
  float* q_f32   = aF32((size_t)B * H2);
  float* scores  = aF32((size_t)B * SE);
  float* aw      = aF32((size_t)B * SE);

  // ---- 1) bf16 conversions (gen_W: 262MB -> 131MB, L2-resident thereafter) ----
  launch_conv(stream, inp,        inp_h,   (size_t)B * SE * E);
  launch_conv(stream, outp,       outp_h,  (size_t)B * SD * E);
  launch_conv(stream, eWih_f,     eWih_fh, (size_t)H3 * E);
  launch_conv(stream, eWhh_f,     eWhh_fh, (size_t)H3 * H);
  launch_conv(stream, eWih_b,     eWih_bh, (size_t)H3 * E);
  launch_conv(stream, eWhh_b,     eWhh_bh, (size_t)H3 * H);
  launch_conv(stream, dWih,       dWih_h,  (size_t)G2 * H2);
  launch_conv(stream, dWhh,       dWhh_h,  (size_t)G2 * H2);
  launch_conv(stream, attn_in_W,  ain_h,   (size_t)H2 * H2);
  launch_conv(stream, attn_out_W, aout_h,  (size_t)E * AQ);
  launch_conv(stream, gen_W,      genW_h,  (size_t)C * AQ);

  // ---- 2) zero initial states ----
  zero_f32<<<(B * H) / 256, 256, 0, stream>>>(hf_f32, B * H);
  zero_f32<<<(B * H) / 256, 256, 0, stream>>>(hb_f32, B * H);
  zero_u16<<<(B * H) / 256, 256, 0, stream>>>(hf_h0, B * H);
  zero_u16<<<(B * H) / 256, 256, 0, stream>>>(hb_h0, B * H);
  zero_u16<<<(B * 2 * E) / 256, 256, 0, stream>>>(decin_h, B * 2 * E);  // op0 = 0

  // ---- 3) hoisted encoder input GEMMs: gi = x @ Wih^T + bih, all timesteps ----
  launch_gemm(stream, 4, inp_h, E, eWih_fh, E, ebih_f, gi_f, H3, nullptr, 0,
              B * SE, H3, E, false);
  launch_gemm(stream, 4, inp_h, E, eWih_bh, E, ebih_b, gi_b, H3, nullptr, 0,
              B * SE, H3, E, false);

  // ---- 4) bidirectional recurrence: one fused WMMA+gate kernel per step ----
  {
    unsigned short *hfc = hf_h0, *hfn = hf_h1;
    unsigned short *hbc = hb_h0, *hbn = hb_h1;
    for (int t = 0; t < SE; ++t) {
      gru_hstep<H><<<H / 16, 32, 0, stream>>>(
          hfc, eWhh_fh, ebhh_f, gi_f + (size_t)t * H3, (long)SE * H3,
          hf_f32, hfn, enc_h + (size_t)t * H2, (long)SE * H2);
      int tb = SE - 1 - t;
      gru_hstep<H><<<H / 16, 32, 0, stream>>>(
          hbc, eWhh_bh, ebhh_b, gi_b + (size_t)tb * H3, (long)SE * H3,
          hb_f32, hbn, enc_h + (size_t)tb * H2 + H, (long)SE * H2);
      unsigned short* tmp;
      tmp = hfc; hfc = hfn; hfn = tmp;
      tmp = hbc; hbc = hbn; hbn = tmp;
    }
  }

  // ---- 5) decoder initial hidden = concat(hf, hb) ----
  build_hd<<<(B * H2) / 256, 256, 0, stream>>>(hf_f32, hb_f32, hd_f32, hd_h0);

  // ---- 6) decoder steps ----
  unsigned short* hd_cur = hd_h0;
  unsigned short* hd_nxt = hd_h1;
  for (int t = 0; t < SD; ++t) {
    build_decin<<<(B * E) / 256, 256, 0, stream>>>(outp_h, t, decin_h);
    // gi = decin @ dWih^T + dbih
    launch_gemm(stream, 1, decin_h, 2 * E, dWih_h, H2, dbih, gi_d, G2, nullptr, 0,
                B, G2, H2, false);
    // fused h @ dWhh^T + GRU gates -> hd (ping-pong bf16)
    gru_hstep<H2><<<H2 / 16, 32, 0, stream>>>(
        hd_cur, dWhh_h, dbhh, gi_d, (long)G2, hd_f32, hd_nxt, nullptr, 0);
    // q = h @ attn_in_W^T   (f32 for scores, bf16 into attnq[:, H2:])
    launch_gemm(stream, 1, hd_nxt, H2, ain_h, H2, nullptr, q_f32, H2,
                attnq_h + H2, AQ, B, H2, H2, false);
    // attention
    attn_scores<<<(B * SE) / 256, 256, 0, stream>>>(enc_h, q_f32, scores);
    softmax_attn<<<B, SE, 0, stream>>>(scores, aw,
                                       attnsBase + (size_t)t * B * SE);
    attn_combine<<<(B * H2) / 256, 256, 0, stream>>>(aw, enc_h, attnq_h);
    // op = tanh(attnq @ attn_out_W^T) -> bf16 into decin[:, E:] for next step
    launch_gemm(stream, 1, attnq_h, AQ, aout_h, AQ, nullptr, nullptr, 0,
                decin_h + E, 2 * E, B, E, AQ, true);
    // gen = attnq @ gen_W^T + gen_b -> outputs[t] (B, C), f32 to d_out
    launch_gemm(stream, 2, attnq_h, AQ, genW_h, AQ, gen_b,
                outF + (size_t)t * B * C, C, nullptr, 0, B, C, AQ, false);
    unsigned short* tmp = hd_cur; hd_cur = hd_nxt; hd_nxt = tmp;
  }
}
